// HGAT_8126078124660
// MI455X (gfx1250) — matrix operand order repaired and checked by reading, not compiled, not run on hardware
//
#include <hip/hip_runtime.h>

typedef __attribute__((ext_vector_type(16))) __bf16        v16bf;
typedef __attribute__((ext_vector_type(8)))  float         v8f;
typedef __attribute__((ext_vector_type(4)))  unsigned int  v4u;
typedef __attribute__((ext_vector_type(8)))  unsigned int  v8u;

union FragU { v8u u; v16bf b; };

// ---------------------------------------------------------------------------
// GEMM: H[n,256] = X[n,K] @ W[K,256], bf16 WMMA, f32 accumulate.
// Block = 256 threads = 8 wave32. Block tile = 64 rows x 256 cols.
// Wave (rt,cg): rt = wave>>1 in 0..3 -> rows [16rt,16rt+16);
//               cg = wave&1          -> cols [128cg,128cg+128) = 8 col tiles.
// Per K-step per wave: 1 A fragment (2x ds_load_b128), 8 B fragments
// (2x ds_load_b128 each from the transposed bf16 tile), 8 WMMAs.
// LDS tiles are bf16, packed into u32 pairs at stage time (v_cvt_pk_bf16_f32).
// Out-of-range rows are handled by CLAMPING the staged row index (no
// divergent guards; clamped rows feed only outputs that the store guard
// discards). Requires K % 32 == 0 (K in {64,128,256} here).
// ---------------------------------------------------------------------------
__global__ __launch_bounds__(256) void gat_gemm_bf16(
    const float* __restrict__ X, const float* __restrict__ W,
    float* __restrict__ H, int K, int n)
{
    // pitch 24 dwords: fragment base offsets stay 16B (A) / 32B (B) aligned
    __shared__ __align__(32) unsigned int AsU[64][24];   // A tile 64x32 bf16
    __shared__ __align__(32) unsigned int BtU[256][24];  // B transposed: [col][K] bf16

    const int tid     = threadIdx.x;
    const int lane    = tid & 31;
    const int wave    = tid >> 5;
    const int rt      = wave >> 1;     // row-tile 0..3
    const int cg      = wave & 1;      // col-half 0..1
    const int rowBase = blockIdx.x * 64;
    const int m       = lane & 15;
    const int kh      = lane >> 4;     // K-half selector

    v8f acc[8];
    #pragma unroll
    for (int t = 0; t < 8; ++t) acc[t] = (v8f){};

    for (int k0 = 0; k0 < K; k0 += 32) {
        // ---- stage A: 64 rows x 16 packed words; 4 words per thread ----
        {
            int idx = tid;
            #pragma unroll
            for (int i = 0; i < 4; ++i) {
                int row   = idx >> 4;        // 0..63
                int kp    = idx & 15;        // packed word -> K pair (2kp,2kp+1)
                int rsafe = rowBase + row;   // clamp instead of branch:
                if (rsafe > n - 1) rsafe = n - 1;
                const float* p = X + (size_t)rsafe * K + k0 + 2 * kp;
                union { __bf16 h[2]; unsigned int u; } pk;
                pk.h[0] = (__bf16)p[0];
                pk.h[1] = (__bf16)p[1];
                AsU[row][kp] = pk.u;
                idx += 256;
            }
        }
        // ---- stage B transposed: thread tid owns column c=tid ----
        {
            const int c = tid;
            #pragma unroll
            for (int p = 0; p < 16; ++p) {
                float w0 = W[(size_t)(k0 + 2 * p) * 256 + c];
                float w1 = W[(size_t)(k0 + 2 * p + 1) * 256 + c];
                union { __bf16 h[2]; unsigned int u; } pk;
                pk.h[0] = (__bf16)w0;
                pk.h[1] = (__bf16)w1;
                BtU[c][p] = pk.u;
            }
        }
        __syncthreads();

        // ---- A fragment: lane kh-half takes K 16kh..16kh+15 ----
        FragU af;
        {
            const int ar = rt * 16 + m;
            v4u a0 = *(const v4u*)&AsU[ar][8 * kh];
            v4u a1 = *(const v4u*)&AsU[ar][8 * kh + 4];
            af.u = (v8u){a0.x, a0.y, a0.z, a0.w, a1.x, a1.y, a1.z, a1.w};
        }
        // ---- 8 B fragments + 8 WMMA, A reused ----
        #pragma unroll
        for (int ct = 0; ct < 8; ++ct) {
            const int col = cg * 128 + ct * 16 + m;
            FragU bf;
            bf.u = *(const v8u*)&BtU[col][8 * kh];   // K = 16kh..16kh+15
            acc[ct] = __builtin_amdgcn_wmma_f32_16x16x32_bf16(
                          false, af.b, false, bf.b, (short)0, acc[ct],
                          false, false);
        }
        __syncthreads();
    }

    // ---- store: VGPR r -> row r + 8*kh within tile, lane%16 -> column ----
    #pragma unroll
    for (int ct = 0; ct < 8; ++ct) {
        const int col = cg * 128 + ct * 16 + m;
        #pragma unroll
        for (int r = 0; r < 8; ++r) {
            int row = rowBase + rt * 16 + r + 8 * kh;
            if (row < n) H[(size_t)row * 256 + col] = acc[ct][r];
        }
    }
}

// ---------------------------------------------------------------------------
// alpha_src/alpha_dst: per (node, head) dot of h[n,h,:] with a_src/a_dst[h,:]
// ---------------------------------------------------------------------------
__global__ __launch_bounds__(256) void gat_alpha(
    const float* __restrict__ H, const float* __restrict__ a_src,
    const float* __restrict__ a_dst, float* __restrict__ al_s,
    float* __restrict__ al_d, int n)
{
    int i = blockIdx.x * 256 + threadIdx.x;
    if (i >= n * 4) return;
    int node = i >> 2, h = i & 3;
    const float* hp = H + (size_t)node * 256 + h * 64;
    const float* as = a_src + h * 64;
    const float* ad = a_dst + h * 64;
    float s = 0.f, d = 0.f;
    #pragma unroll 4
    for (int c = 0; c < 64; ++c) {
        float v = hp[c];
        s += v * as[c];
        d += v * ad[c];
    }
    al_s[i] = s;
    al_d[i] = d;
}

// ---------------------------------------------------------------------------
// softmax state init + bias init (re-run every call; ws is poisoned)
// ---------------------------------------------------------------------------
__global__ void gat_init_softmax(unsigned int* __restrict__ emax,
                                 float* __restrict__ denom, int n4)
{
    int i = blockIdx.x * 256 + threadIdx.x;
    if (i < n4) { emax[i] = 0u; denom[i] = 0.f; }
}

__global__ void gat_init_out(float* __restrict__ out,
                             const float* __restrict__ b, int total)
{
    int i = blockIdx.x * 256 + threadIdx.x;
    if (i < total) out[i] = b[i & 255];
}

// ---------------------------------------------------------------------------
// Edge passes. Order-preserving uint encoding so atomicMax(u32) == float max.
// ---------------------------------------------------------------------------
__device__ __forceinline__ float lrelu(float x) { return x > 0.f ? x : 0.2f * x; }
__device__ __forceinline__ unsigned int ordEnc(float f) {
    unsigned int u = __float_as_uint(f);
    return (u & 0x80000000u) ? ~u : (u | 0x80000000u);
}
__device__ __forceinline__ float ordDec(unsigned int o) {
    return __uint_as_float((o & 0x80000000u) ? (o ^ 0x80000000u) : ~o);
}

__global__ __launch_bounds__(256) void gat_edge_max(
    const int* __restrict__ ei, int E, int n,
    const float* __restrict__ al_s, const float* __restrict__ al_d,
    unsigned int* __restrict__ emax)
{
    int i = blockIdx.x * 256 + threadIdx.x;
    if (i >= E + n) return;
    int s = (i < E) ? ei[i]     : (i - E);
    int d = (i < E) ? ei[E + i] : (i - E);
    #pragma unroll
    for (int h = 0; h < 4; ++h) {
        float e = lrelu(al_s[s * 4 + h] + al_d[d * 4 + h]);
        atomicMax(&emax[d * 4 + h], ordEnc(e));
    }
}

__global__ __launch_bounds__(256) void gat_edge_denom(
    const int* __restrict__ ei, int E, int n,
    const float* __restrict__ al_s, const float* __restrict__ al_d,
    const unsigned int* __restrict__ emax, float* __restrict__ denom)
{
    int i = blockIdx.x * 256 + threadIdx.x;
    if (i >= E + n) return;
    int s = (i < E) ? ei[i]     : (i - E);
    int d = (i < E) ? ei[E + i] : (i - E);
    #pragma unroll
    for (int h = 0; h < 4; ++h) {
        float e  = lrelu(al_s[s * 4 + h] + al_d[d * 4 + h]);
        float ex = __expf(e - ordDec(emax[d * 4 + h]));
        atomicAdd(&denom[d * 4 + h], ex);
    }
}

// 64 lanes per edge (one per channel); gathers/scatters should hit the 192MB L2
__global__ __launch_bounds__(256) void gat_edge_agg(
    const int* __restrict__ ei, int E, int n,
    const float* __restrict__ al_s, const float* __restrict__ al_d,
    const unsigned int* __restrict__ emax, const float* __restrict__ denom,
    const float* __restrict__ H, float* __restrict__ out)
{
    int gid = blockIdx.x * 256 + threadIdx.x;
    int i = gid >> 6;   // edge id
    int c = gid & 63;   // channel
    if (i >= E + n) return;
    int s = (i < E) ? ei[i]     : (i - E);
    int d = (i < E) ? ei[E + i] : (i - E);
    #pragma unroll
    for (int h = 0; h < 4; ++h) {
        float e     = lrelu(al_s[s * 4 + h] + al_d[d * 4 + h]);
        float ex    = __expf(e - ordDec(emax[d * 4 + h]));
        float alpha = ex / (denom[d * 4 + h] + 1e-16f);
        atomicAdd(&out[(size_t)d * 256 + h * 64 + c],
                  alpha * H[(size_t)s * 256 + h * 64 + c]);
    }
}

// ---------------------------------------------------------------------------
// One full GATConv: GEMM -> alphas -> softmax(max/denom) -> aggregate (+bias)
// ---------------------------------------------------------------------------
static void run_conv(const float* X, int K, const int* ei, int E,
                     const float* W, const float* a_src, const float* a_dst,
                     const float* bias, float* Hbuf, float* out,
                     float* al_s, float* al_d, unsigned int* emax, float* denom,
                     int n, hipStream_t stream)
{
    gat_gemm_bf16<<<(n + 63) / 64, 256, 0, stream>>>(X, W, Hbuf, K, n);
    int n4 = n * 4;
    gat_alpha<<<(n4 + 255) / 256, 256, 0, stream>>>(Hbuf, a_src, a_dst, al_s, al_d, n);
    gat_init_softmax<<<(n4 + 255) / 256, 256, 0, stream>>>(emax, denom, n4);
    gat_init_out<<<(n * 256 + 255) / 256, 256, 0, stream>>>(out, bias, n * 256);
    int EP = E + n;
    gat_edge_max  <<<(EP + 255) / 256, 256, 0, stream>>>(ei, E, n, al_s, al_d, emax);
    gat_edge_denom<<<(EP + 255) / 256, 256, 0, stream>>>(ei, E, n, al_s, al_d, emax, denom);
    int tot = EP * 64;
    gat_edge_agg  <<<(tot + 255) / 256, 256, 0, stream>>>(ei, E, n, al_s, al_d, emax,
                                                          denom, Hbuf, out);
}

extern "C" void kernel_launch(void* const* d_in, const int* in_sizes, int n_in,
                              void* d_out, int out_size, void* d_ws, size_t ws_size,
                              hipStream_t stream)
{
    const float* x_user = (const float*)d_in[0];
    const float* x_post = (const float*)d_in[1];
    const int*   ei_u   = (const int*)d_in[2];
    const int*   ei_p   = (const int*)d_in[3];
    const float* W1u  = (const float*)d_in[4];
    const float* a1us = (const float*)d_in[5];
    const float* a1ud = (const float*)d_in[6];
    const float* b1u  = (const float*)d_in[7];
    const float* W1p  = (const float*)d_in[8];
    const float* a1ps = (const float*)d_in[9];
    const float* a1pd = (const float*)d_in[10];
    const float* b1p  = (const float*)d_in[11];
    const float* W2u  = (const float*)d_in[12];
    const float* a2us = (const float*)d_in[13];
    const float* a2ud = (const float*)d_in[14];
    const float* b2u  = (const float*)d_in[15];
    const float* W2p  = (const float*)d_in[16];
    const float* a2ps = (const float*)d_in[17];
    const float* a2pd = (const float*)d_in[18];
    const float* b2p  = (const float*)d_in[19];

    const int N  = in_sizes[0] / 128;   // 50000
    const int Kp = in_sizes[1] / N;     // 64
    const int E  = in_sizes[2] / 2;     // 800000

    // workspace layout (floats)
    float* wsf = (float*)d_ws;
    size_t nhc = (size_t)N * 256;
    float* Hbuf = wsf;                 // pre-aggregation features (per conv)
    float* hu   = wsf + nhc;           // layer-1 user output
    float* hp   = wsf + 2 * nhc;       // layer-1 post output
    float* al_s = wsf + 3 * nhc;
    float* al_d = al_s + (size_t)N * 4;
    unsigned int* emax = (unsigned int*)(al_d + (size_t)N * 4);
    float* denom = (float*)(emax + (size_t)N * 4);

    float* ou = (float*)d_out;
    float* op = ou + nhc;

    // layer 1
    run_conv(x_user, 128, ei_u, E, W1u, a1us, a1ud, b1u, Hbuf, hu,
             al_s, al_d, emax, denom, N, stream);
    run_conv(x_post, Kp,  ei_p, E, W1p, a1ps, a1pd, b1p, Hbuf, hp,
             al_s, al_d, emax, denom, N, stream);
    // layer 2
    run_conv(hu, 256, ei_u, E, W2u, a2us, a2ud, b2u, Hbuf, ou,
             al_s, al_d, emax, denom, N, stream);
    run_conv(hp, 256, ei_p, E, W2p, a2ps, a2pd, b2p, Hbuf, op,
             al_s, al_d, emax, denom, N, stream);
}
